// MultiScaleResidualQuantizer3D_80590766342854
// MI455X (gfx1250) — matrix operand order repaired and checked by reading, not compiled, not run on hardware
//
#include <hip/hip_runtime.h>
#include <hip/hip_bf16.h>
#include <math.h>

#define HW   16
#define CCH  32
#define NB   128
#define NE   4096
#define NELEM (NB*CCH*HW*HW)   // 1,048,576

typedef __attribute__((ext_vector_type(16))) __bf16 v16bf;
typedef __attribute__((ext_vector_type(8)))  float  v8f;

__device__ __forceinline__ float wave_sum32(float v) {
  #pragma unroll
  for (int m = 1; m < 32; m <<= 1) v += __shfl_xor(v, m, 32);
  return v;
}

// ---------------- init: zero f_hat, f_rest = f_input, zero hist/loss ----------------
__global__ void k_init(const float* __restrict__ fin, float* __restrict__ fhat,
                       float* __restrict__ frest, int* __restrict__ hist,
                       float* __restrict__ lacc) {
  int i = blockIdx.x * blockDim.x + threadIdx.x;
  if (i < NELEM) { fhat[i] = 0.f; frest[i] = fin[i]; }
  if (i < NE) hist[i] = 0;
  if (i == 0) lacc[0] = 0.f;
}

// ---------------- normalize codebook rows -> bf16 [NE][32] (row = B operand) -------
__global__ void k_cbnorm(const float* __restrict__ cb, unsigned short* __restrict__ cbb) {
  int i = blockIdx.x * blockDim.x + threadIdx.x;
  if (i >= NE) return;
  const float* r = cb + (size_t)i * CCH;
  float s = 0.f;
  #pragma unroll
  for (int c = 0; c < CCH; ++c) s += r[c] * r[c];
  float inv = 1.f / fmaxf(sqrtf(s), 1e-12f);
  __bf16* o = (__bf16*)(cbb + (size_t)i * CCH);
  #pragma unroll
  for (int c = 0; c < CCH; ++c) o[c] = (__bf16)(r[c] * inv);
}

// ---------------- area-downsample + L2-normalize tokens (bf16) ----------------------
// one wave32 per token, lane = channel
__global__ void k_tokens(const float* __restrict__ frest, unsigned short* __restrict__ tok,
                         int pn) {
  int wid = blockIdx.x * (blockDim.x >> 5) + (threadIdx.x >> 5);
  int c   = threadIdx.x & 31;
  int pp  = pn * pn;
  int b   = wid / pp;
  int r   = wid - b * pp;
  int o   = r / pn, p = r - o * pn;
  int ys = (o * HW) / pn, ye = ((o + 1) * HW + pn - 1) / pn;  // area pooling ranges
  int xs = (p * HW) / pn, xe = ((p + 1) * HW + pn - 1) / pn;
  const float* base = frest + ((size_t)(b * CCH + c)) * HW * HW;
  float s = 0.f;
  for (int y = ys; y < ye; ++y)
    for (int x = xs; x < xe; ++x) s += base[y * HW + x];
  float v = s / (float)((ye - ys) * (xe - xs));
  float sq  = wave_sum32(v * v);
  float inv = 1.f / fmaxf(sqrtf(sq), 1e-12f);
  ((__bf16*)tok)[(size_t)wid * CCH + c] = (__bf16)(v * inv);
}

// ---------------- WMMA cosine-similarity argmax over 4096 codes ---------------------
// 4 waves/WG, 2 token-tiles (32 tokens) per wave -> each LDS B-tile feeds two
// independent WMMAs. Codebook staged in 64KB LDS slices via async global->LDS DMA.
__device__ __forceinline__ void upd8(float (&best)[8], int (&bidx)[8], v8f acc, int cand) {
  #pragma unroll
  for (int r = 0; r < 8; ++r)
    if (acc[r] > best[r]) { best[r] = acc[r]; bidx[r] = cand; }
}

__device__ __forceinline__ void reduce_write(float (&best)[8], int (&bidx)[8],
                                             int tile, int half, int l15,
                                             int* __restrict__ idx) {
  #pragma unroll
  for (int r = 0; r < 8; ++r) {
    float bv = best[r]; int bi = bidx[r];
    #pragma unroll
    for (int off = 1; off < 16; off <<= 1) {
      float ov = __shfl_xor(bv, off, 32);
      int   oi = __shfl_xor(bi, off, 32);
      if (ov > bv || (ov == bv && oi < bi)) { bv = ov; bi = oi; }  // ties -> lowest idx
    }
    if (l15 == 0) idx[tile * 16 + r + 8 * half] = bi;
  }
}

__global__ void __launch_bounds__(128)
k_argmax(const unsigned short* __restrict__ tok, const unsigned short* __restrict__ cbb,
         int* __restrict__ idx) {
  extern __shared__ uint4 lds[];                 // 4096 * 16B = 64KB slice
  const int tid   = threadIdx.x;
  const int lane  = tid & 31;
  const int wave  = tid >> 5;
  const int tile0 = blockIdx.x * 8 + wave * 2;   // two 16-token tiles per wave
  const int half  = lane >> 4;                   // ISA half-wave select
  const int l15   = lane & 15;

  // A operands: 16x32 bf16, ISA layout: lanes0-15 row M=l15 {K0..7,K16..23},
  // lanes16-31 row M=l15 {K8..15,K24..31}
  union Pack { uint4 u[2]; v16bf v; };
  Pack A0, A1;
  {
    const uint4* a0 = (const uint4*)(tok + (size_t)(tile0 * 16 + l15) * CCH);
    const uint4* a1 = (const uint4*)(tok + (size_t)((tile0 + 1) * 16 + l15) * CCH);
    A0.u[0] = a0[half]; A0.u[1] = a0[2 + half];
    A1.u[0] = a1[half]; A1.u[1] = a1[2 + half];
  }

  float best0[8], best1[8];
  int   bidx0[8], bidx1[8];
  #pragma unroll
  for (int r = 0; r < 8; ++r) {
    best0[r] = -3.4e38f; bidx0[r] = 0;
    best1[r] = -3.4e38f; bidx1[r] = 0;
  }

  for (int slice = 0; slice < 4; ++slice) {
    __syncthreads();
    const uint4* src = (const uint4*)(cbb + (size_t)slice * 1024 * CCH);
    // async DMA: each lane copies 16B global -> LDS (ASYNCcnt-tracked)
    for (int i = tid; i < 4096; i += 128) {
      asm volatile("global_load_async_to_lds_b128 %0, %1, off"
                   :: "v"((unsigned)(size_t)(lds + i)), "v"(src + i)
                   : "memory");
    }
    if (slice < 3)
      __builtin_prefetch(cbb + (size_t)(slice + 1) * 1024 * CCH, 0, 1); // global_prefetch_b8
#if __has_builtin(__builtin_amdgcn_s_wait_asynccnt)
    __builtin_amdgcn_s_wait_asynccnt(0);
#else
    asm volatile("s_wait_asynccnt 0x0" ::: "memory");
#endif
    __syncthreads();

    #pragma unroll 4
    for (int nt = 0; nt < 64; ++nt) {
      int n = nt * 16 + l15;                     // code row in slice (B column N=l15)
      Pack B;
      B.u[0] = lds[(n << 2) + half];
      B.u[1] = lds[(n << 2) + 2 + half];
      v8f acc0 = {};
      acc0 = __builtin_amdgcn_wmma_f32_16x16x32_bf16(
          false, A0.v, false, B.v, (short)0, acc0, false, false);
      v8f acc1 = {};
      acc1 = __builtin_amdgcn_wmma_f32_16x16x32_bf16(
          false, A1.v, false, B.v, (short)0, acc1, false, false);
      int cand = slice * 1024 + n;
      upd8(best0, bidx0, acc0, cand);
      upd8(best1, bidx1, acc1, cand);
    }
  }
  reduce_write(best0, bidx0, tile0,     half, l15, idx);
  reduce_write(best1, bidx1, tile0 + 1, half, l15, idx);
}

// ---------------- gather + bicubic upsample (identity at pn==16) --------------------
__device__ __forceinline__ float cubw(float t) {
  const float a = -0.75f;
  t = fabsf(t);
  if (t <= 1.f) return ((a + 2.f) * t - (a + 3.f)) * t * t + 1.f;
  if (t < 2.f)  return (((t - 5.f) * t + 8.f) * t - 4.f) * a;
  return 0.f;
}

__global__ void k_gather_up(const int* __restrict__ idx, const float* __restrict__ cb,
                            float* __restrict__ h, int pn) {
  int t = blockIdx.x * blockDim.x + threadIdx.x;  // [b][c][y][x]
  int x = t & 15, y = (t >> 4) & 15;
  int c = (t >> 8) & 31;
  int b = t >> 13;
  float scale = (float)pn / 16.f;
  float sy = (y + 0.5f) * scale - 0.5f;
  float sx = (x + 0.5f) * scale - 0.5f;
  int iy = (int)floorf(sy), ix = (int)floorf(sx);
  float fy = sy - iy, fx = sx - ix;
  float wy[4], wx[4]; int jy[4], jx[4];
  #pragma unroll
  for (int o = 0; o < 4; ++o) {
    wy[o] = cubw(fy - (float)(o - 1));
    wx[o] = cubw(fx - (float)(o - 1));
    jy[o] = min(max(iy + o - 1, 0), pn - 1);     // border replicate
    jx[o] = min(max(ix + o - 1, 0), pn - 1);
  }
  const int* ib = idx + b * pn * pn;
  float s = 0.f;
  #pragma unroll
  for (int o = 0; o < 4; ++o) {
    float rs = 0.f;
    #pragma unroll
    for (int p = 0; p < 4; ++p)
      rs = fmaf(wx[p], cb[(size_t)ib[jy[o] * pn + jx[p]] * CCH + c], rs);
    s = fmaf(wy[o], rs, s);
  }
  h[t] = s;
}

// ---------------- phi conv (fp32) + residual mix + state update + loss --------------
__global__ void __launch_bounds__(256)
k_phi(const float* __restrict__ h, const float* __restrict__ W,
      const float* __restrict__ bias, const float* __restrict__ fin,
      float* __restrict__ fhat, float* __restrict__ frest, float* __restrict__ lacc) {
  extern __shared__ float sm[];
  float* hs  = sm;           // 8192  : one image [32][16][16]
  float* ws_ = sm + 8192;    // 9216  : weights [co][ci][9]
  float* bs  = ws_ + 9216;   // 32
  float* red = bs + 32;      // 256
  const int tid = threadIdx.x;
  const int b   = blockIdx.x;
  const float* hb = h + (size_t)b * 8192;
  for (int i = tid; i < 8192; i += 256) hs[i]  = hb[i];
  for (int i = tid; i < 9216; i += 256) ws_[i] = W[i];
  if (tid < 32) bs[tid] = bias[tid];
  __syncthreads();
  const int y = tid >> 4, x = tid & 15;
  float acc[32];
  #pragma unroll
  for (int co = 0; co < 32; ++co) acc[co] = 0.f;
  for (int ci = 0; ci < 32; ++ci) {
    float v[9];
    #pragma unroll
    for (int dy = -1; dy <= 1; ++dy)
      #pragma unroll
      for (int dx = -1; dx <= 1; ++dx) {
        int yy = y + dy, xx = x + dx;
        v[(dy + 1) * 3 + dx + 1] =
            ((unsigned)yy < 16u && (unsigned)xx < 16u) ? hs[(ci << 8) + (yy << 4) + xx] : 0.f;
      }
    const float* wr = ws_ + ci * 9;
    #pragma unroll
    for (int co = 0; co < 32; ++co) {
      const float* wc = wr + co * 288;
      float a2 = acc[co];
      #pragma unroll
      for (int j = 0; j < 9; ++j) a2 = fmaf(wc[j], v[j], a2);
      acc[co] = a2;
    }
  }
  float lsq = 0.f;
  #pragma unroll
  for (int co = 0; co < 32; ++co) {
    float hv = hs[(co << 8) + tid];
    float h2 = 0.5f * hv + 0.5f * (acc[co] + bs[co]);  // RESI = 0.5
    size_t off = ((size_t)(b * 32 + co)) * 256 + tid;
    float fh = fhat[off] + h2;
    fhat[off]  = fh;
    frest[off] -= h2;
    float d = fh - fin[off];
    lsq = fmaf(d, d, lsq);
  }
  red[tid] = lsq;
  __syncthreads();
  for (int s2 = 128; s2 > 0; s2 >>= 1) {
    if (tid < s2) red[tid] += red[tid + s2];
    __syncthreads();
  }
  if (tid == 0) atomicAdd(lacc, red[0]);
}

// ---------------- last-scale code histogram + scalars -------------------------------
__global__ void k_hist(const int* __restrict__ idx, int* __restrict__ hist, int n) {
  int i = blockIdx.x * blockDim.x + threadIdx.x;
  if (i < n) atomicAdd(&hist[idx[i]], 1);
}

__global__ void k_final(const int* __restrict__ hist, const float* __restrict__ lacc,
                        float* __restrict__ out) {
  __shared__ float red[256];
  int tid = threadIdx.x;
  float s = 0.f;
  for (int i = tid; i < NE; i += 256) {
    float p = (float)hist[i] * (1.f / 32768.f);
    s += p * logf(p + 1e-10f);
  }
  red[tid] = s;
  __syncthreads();
  for (int k = 128; k > 0; k >>= 1) { if (tid < k) red[tid] += red[tid + k]; __syncthreads(); }
  if (tid == 0) {
    out[NELEM]     = lacc[0] * (1.25f / (10.f * (float)NELEM));  // (BETA+1)*mean / SN
    out[NELEM + 1] = expf(-red[0]);
  }
}

extern "C" void kernel_launch(void* const* d_in, const int* in_sizes, int n_in,
                              void* d_out, int out_size, void* d_ws, size_t ws_size,
                              hipStream_t stream) {
  (void)in_sizes; (void)n_in; (void)out_size; (void)ws_size;
  const float* fin = (const float*)d_in[0];
  const float* cb  = (const float*)d_in[1];
  const float* pw  = (const float*)d_in[2];
  const float* pb  = (const float*)d_in[3];
  float* out  = (float*)d_out;
  float* fhat = out;  // accumulate f_hat directly in output

  char* ws = (char*)d_ws;
  size_t off = 0;
  float*          frest = (float*)(ws + off);          off += (size_t)NELEM * 4;          // 4MB
  unsigned short* cbb   = (unsigned short*)(ws + off); off += (size_t)NE * CCH * 2;       // 256KB
  unsigned short* tok   = (unsigned short*)(ws + off); off += (size_t)NB * 256 * CCH * 2; // 2MB
  int*            idxb  = (int*)(ws + off);            off += (size_t)NB * 256 * 4;       // 128KB
  float*          hbuf  = (float*)(ws + off);          off += (size_t)NELEM * 4;          // 4MB
  int*            hist  = (int*)(ws + off);            off += (size_t)NE * 4;
  float*          lacc  = (float*)(ws + off);          off += 16;

  k_init<<<4096, 256, 0, stream>>>(fin, fhat, frest, hist, lacc);
  k_cbnorm<<<16, 256, 0, stream>>>(cb, cbb);

  static const int PNSl[10] = {1, 2, 3, 4, 5, 6, 8, 10, 13, 16};
  static const int PHIk[10] = {0, 0, 0, 1, 1, 2, 2, 2, 3, 3};  // PhiPartiallyShared ticks

  for (int si = 0; si < 10; ++si) {
    int pn = PNSl[si];
    int T  = NB * pn * pn;                       // tokens this scale (multiple of 128)
    k_tokens<<<T / 4, 128, 0, stream>>>(frest, tok, pn);
    k_argmax<<<T / 128, 128, 65536, stream>>>(tok, cbb, idxb);
    k_gather_up<<<4096, 256, 0, stream>>>(idxb, cb, hbuf, pn);
    k_phi<<<NB, 256, 70784, stream>>>(hbuf, pw + (size_t)PHIk[si] * 32 * 32 * 9,
                                      pb + (size_t)PHIk[si] * 32, fin, fhat, frest, lacc);
    if (si == 9) k_hist<<<T / 256, 256, 0, stream>>>(idxb, hist, T);
  }
  k_final<<<1, 256, 0, stream>>>(hist, lacc, out);
}
